// GumbelSinkhorn_87308095193258
// MI455X (gfx1250) — 8-bit matrix operand order repaired and checked by reading, compiled, not run on hardware
//
#include <hip/hip_runtime.h>

typedef __attribute__((ext_vector_type(8))) int   v8i;
typedef __attribute__((ext_vector_type(8))) float v8f;

#define N_DIM    512
#define NTHREADS 1024
#define ITERS    60
#define INV_TEMP (1.0f / 3.0f)
#define GAMMA_C  0.01f
#define SCALE_S  512.0f

__device__ __forceinline__ unsigned int fp8_encode(float x) {
  x = fminf(x, 448.0f);
  if (x < 0.015625f) {
    return (unsigned int)__float2int_rn(x * 512.0f);
  }
  unsigned int b = __float_as_uint(x);
  int ex = (int)((b >> 23) & 0xFF) - 127;
  float scale = __uint_as_float((unsigned int)((3 - ex + 127) << 23));
  int r = __float2int_rn(x * scale);
  if (r == 16) { ex += 1; r = 8; }
  return (unsigned int)(((ex + 7) << 3) | (r - 8));
}

__device__ __forceinline__ unsigned int fp8_encode_s(float x) {
  unsigned int s = (__float_as_uint(x) >> 31) << 7;
  return s | fp8_encode(fabsf(x));
}

__device__ __forceinline__ float fp8_decode(unsigned int h) {
  unsigned int e = (h >> 3) & 0xF;
  unsigned int m = h & 7;
  if (e == 0) return (float)m * 0.001953125f;
  return __uint_as_float(((e + 120u) << 23) | (m << 20));
}

__device__ __forceinline__ float kexp(float lg, float uu) {
  float eps = -__logf(fmaxf(-__logf(uu), 1e-30f));
  return __expf((lg + GAMMA_C * eps) * INV_TEMP);
}

__device__ __forceinline__ int joffA(int v, int g) {
  return (g << 3) + ((v >> 2) << 5) + (((v >> 1) & 1) << 4) + ((v & 1) << 2);
}
__device__ __forceinline__ int koffB(int v, int g) {
  return (g << 3) + ((v >> 2) << 5) + (((v >> 1) & 1) << 4) + ((v & 1) << 2);
}

__global__ __launch_bounds__(NTHREADS)
void gumbel_sinkhorn_fp8_wmma(const float* __restrict__ logits,
                              const float* __restrict__ u,
                              float* __restrict__ out)
{
  __shared__ float r_f[N_DIM];
  __shared__ float c_f[N_DIM];
  __shared__ __align__(4) unsigned char rhi[N_DIM];
  __shared__ __align__(4) unsigned char rlo[N_DIM];
  __shared__ __align__(4) unsigned char chi[N_DIM];
  __shared__ __align__(4) unsigned char clo[N_DIM];

  const int bat  = blockIdx.x;
  const int tid  = threadIdx.x;
  const int wave = tid >> 5;
  const int lane = tid & 31;
  const int grp  = lane >> 4;
  const int ln   = lane & 15;

  const float* L = logits + (size_t)bat * N_DIM * N_DIM;
  const float* U = u      + (size_t)bat * N_DIM * N_DIM;
  float*       O = out    + (size_t)bat * N_DIM * N_DIM;

  for (int i = tid; i < N_DIM; i += NTHREADS) { chi[i] = 0x38; clo[i] = 0; }

  v8i Af[8];
  v8i Bf[8];
  const int rowA = (wave << 4) + ln;
  const int colB = (wave << 4) + ln;

  #pragma unroll
  for (int cb = 0; cb < 8; ++cb) {
    #pragma unroll
    for (int v = 0; v < 8; ++v) {
      const int j0 = (cb << 6) + joffA(v, grp);
      const float4 l4 = *(const float4*)(L + (size_t)rowA * N_DIM + j0);
      const float4 u4 = *(const float4*)(U + (size_t)rowA * N_DIM + j0);
      unsigned int d =  fp8_encode(kexp(l4.x, u4.x))
                     | (fp8_encode(kexp(l4.y, u4.y)) << 8)
                     | (fp8_encode(kexp(l4.z, u4.z)) << 16)
                     | (fp8_encode(kexp(l4.w, u4.w)) << 24);
      Af[cb][v] = (int)d;
    }
  }
  #pragma unroll
  for (int rb = 0; rb < 8; ++rb) {
    #pragma unroll
    for (int v = 0; v < 8; ++v) {
      const int k0 = (rb << 6) + koffB(v, grp);
      unsigned int d = 0;
      #pragma unroll
      for (int bb = 0; bb < 4; ++bb) {
        const size_t idx = (size_t)(k0 + bb) * N_DIM + colB;
        d |= fp8_encode(kexp(L[idx], U[idx])) << (8 * bb);
      }
      Bf[rb][v] = (int)d;
    }
  }
  __syncthreads();

  #pragma unroll 1
  for (int it = 0; it <= ITERS; ++it) {
    {
      v8f D = {0.f, 0.f, 0.f, 0.f, 0.f, 0.f, 0.f, 0.f};
      #pragma unroll
      for (int cb = 0; cb < 8; ++cb) {
        v8i Bh, Bl;
        #pragma unroll
        for (int v = 0; v < 8; ++v) {
          const int k = (cb << 6) + koffB(v, grp);
          Bh[v] = *(const int*)&chi[k];
          Bl[v] = *(const int*)&clo[k];
        }
        D = __builtin_amdgcn_wmma_f32_16x16x64_fp8_fp8(Af[cb], Bh, (short)0, D, false, false);
        D = __builtin_amdgcn_wmma_f32_16x16x64_fp8_fp8(Af[cb], Bl, (short)0, D, false, false);
      }
      if (ln == 0) {
        #pragma unroll
        for (int d = 0; d < 8; ++d) {
          const int m = (wave << 4) + (grp << 3) + d;
          const float inv = SCALE_S / D[d];
          r_f[m] = inv;
          const unsigned int h = fp8_encode(inv);
          rhi[m] = (unsigned char)h;
          rlo[m] = (unsigned char)fp8_encode_s(inv - fp8_decode(h));
        }
      }
    }
    __syncthreads();
    if (it == ITERS) break;

    {
      v8f D = {0.f, 0.f, 0.f, 0.f, 0.f, 0.f, 0.f, 0.f};
      #pragma unroll
      for (int rb = 0; rb < 8; ++rb) {
        v8i Ah, Al;
        #pragma unroll
        for (int v = 0; v < 8; ++v) {
          const int k = (rb << 6) + joffA(v, grp);
          Ah[v] = *(const int*)&rhi[k];
          Al[v] = *(const int*)&rlo[k];
        }
        D = __builtin_amdgcn_wmma_f32_16x16x64_fp8_fp8(Ah, Bf[rb], (short)0, D, false, false);
        D = __builtin_amdgcn_wmma_f32_16x16x64_fp8_fp8(Al, Bf[rb], (short)0, D, false, false);
      }
      if (grp == 0) {
        const int j = (wave << 4) + ln;
        const float inv = SCALE_S / D[0];
        c_f[j] = inv;
        const unsigned int h = fp8_encode(inv);
        chi[j] = (unsigned char)h;
        clo[j] = (unsigned char)fp8_encode_s(inv - fp8_decode(h));
      }
    }
    __syncthreads();
  }

  const float invS = 1.0f / SCALE_S;
  for (int idx = tid; idx < N_DIM * N_DIM; idx += NTHREADS) {
    const int i = idx >> 9;
    const int j = idx & (N_DIM - 1);
    const float kv = kexp(L[idx], U[idx]);
    O[idx] = r_f[i] * kv * c_f[j] * invS;
  }
}

extern "C" void kernel_launch(void* const* d_in, const int* in_sizes, int n_in,
                              void* d_out, int out_size, void* d_ws, size_t ws_size,
                              hipStream_t stream) {
  (void)n_in; (void)d_ws; (void)ws_size;
  const float* logits = (const float*)d_in[0];
  const float* u      = (const float*)d_in[1];
  float* out          = (float*)d_out;
  const int batches   = in_sizes[0] / (N_DIM * N_DIM);
  (void)out_size;
  gumbel_sinkhorn_fp8_wmma<<<batches, NTHREADS, 0, stream>>>(logits, u, out);
}
